// Attention_73744588472673
// MI455X (gfx1250) — matrix operand order repaired and checked by reading, not compiled, not run on hardware
//
#include <hip/hip_runtime.h>
#include <hip/hip_bf16.h>

typedef _Float16 v16h __attribute__((ext_vector_type(16)));
typedef _Float16 h8   __attribute__((ext_vector_type(8)));
typedef _Float16 h4   __attribute__((ext_vector_type(4)));
typedef float    v8f  __attribute__((ext_vector_type(8)));

#define DIM    768
#define HEADS  12
#define HD     64
#define N_TOK  1025
#define BATCH  8
#define NPAD   1056                 // 33 * 32, covers all kv tile reads
#define M_ROWS (BATCH * N_TOK)      // 8200
#define Q_SCALE 0.125f              // 64^-0.5
#define KSTEPS (DIM / 32)           // 24
#define LROW   40                   // padded LDS row stride (halfs)

#define WMMA_F16(a, b, c) \
  __builtin_amdgcn_wmma_f32_16x16x32_f16(false, (a), false, (b), (short)0, (c), false, false)

__device__ __forceinline__ v16h ld16(const _Float16* lo, const _Float16* hi) {
  h8 a = *(const h8*)lo;
  h8 b = *(const h8*)hi;
  v16h r;
#pragma unroll
  for (int i = 0; i < 8; ++i) { r[i] = a[i]; r[i + 8] = b[i]; }
  return r;
}

__device__ __forceinline__ h4 cvt4(float4 v) {
  h4 r; r[0] = (_Float16)v.x; r[1] = (_Float16)v.y;
  r[2] = (_Float16)v.z; r[3] = (_Float16)v.w;
  return r;
}

__device__ __forceinline__ float rmax16(float v) {
  v = fmaxf(v, __shfl_xor(v, 8, 16));
  v = fmaxf(v, __shfl_xor(v, 4, 16));
  v = fmaxf(v, __shfl_xor(v, 2, 16));
  v = fmaxf(v, __shfl_xor(v, 1, 16));
  return v;
}
__device__ __forceinline__ float rsum16(float v) {
  v += __shfl_xor(v, 8, 16);
  v += __shfl_xor(v, 4, 16);
  v += __shfl_xor(v, 2, 16);
  v += __shfl_xor(v, 1, 16);
  return v;
}

// ---------------------------------------------------------------------------
// Kernel 1: qkv = x @ qkv_w^T + [q_bias, 0, v_bias]; split/scale into f16
// Q,K: [B,H,NPAD,64]; V stored transposed: [B,H,64,NPAD].
// Double-buffered LDS; vectorized branch-free staging loads.
// ---------------------------------------------------------------------------
__global__ void __launch_bounds__(256) qkv_kernel(
    const float* __restrict__ x, const float* __restrict__ w,
    const float* __restrict__ qb, const float* __restrict__ vb,
    _Float16* __restrict__ qws, _Float16* __restrict__ kws,
    _Float16* __restrict__ vtws) {
  __shared__ _Float16 la[2][64 * LROW];
  __shared__ _Float16 lb[2][64 * LROW];
  const int m0 = blockIdx.x * 64;
  const int n0 = blockIdx.y * 64;
  const int tid = threadIdx.x;
  const int wave = tid >> 5, lane = tid & 31, grp = lane >> 4, ln = lane & 15;
  const int rs = wave >> 1;         // row subtile 0..3
  const int cs = (wave & 1) * 2;    // col subtiles cs, cs+1

  // staging indices: two float4 per thread per tile
  const int i0 = tid, i1 = tid + 256;
  const int ar0 = i0 >> 3, ac0 = (i0 & 7) * 4;
  const int ar1 = i1 >> 3, ac1 = (i1 & 7) * 4;
  const int gra0 = m0 + ar0, gra1 = m0 + ar1;
  const size_t rowA0 = (size_t)(gra0 < M_ROWS ? gra0 : M_ROWS - 1) * DIM;
  const size_t rowA1 = (size_t)(gra1 < M_ROWS ? gra1 : M_ROWS - 1) * DIM;
  const size_t rowB0 = (size_t)(n0 + ar0) * DIM;
  const size_t rowB1 = (size_t)(n0 + ar1) * DIM;

  float4 fa0, fa1, fb0, fb1;
  auto load_tiles = [&](int kk) {
    fa0 = *(const float4*)(x + rowA0 + kk + ac0);
    fa1 = *(const float4*)(x + rowA1 + kk + ac1);
    fb0 = *(const float4*)(w + rowB0 + kk + ac0);
    fb1 = *(const float4*)(w + rowB1 + kk + ac1);
    if (gra0 >= M_ROWS) fa0 = make_float4(0.f, 0.f, 0.f, 0.f);
    if (gra1 >= M_ROWS) fa1 = make_float4(0.f, 0.f, 0.f, 0.f);
  };
  auto store_tiles = [&](int buf) {
    *(h4*)&la[buf][ar0 * LROW + ac0] = cvt4(fa0);
    *(h4*)&la[buf][ar1 * LROW + ac1] = cvt4(fa1);
    *(h4*)&lb[buf][ar0 * LROW + ac0] = cvt4(fb0);
    *(h4*)&lb[buf][ar1 * LROW + ac1] = cvt4(fb1);
  };

  v8f c0 = {}, c1 = {};
  load_tiles(0);
  store_tiles(0);
  __syncthreads();
  for (int ks = 0; ks < KSTEPS; ++ks) {
    if (ks + 1 < KSTEPS) load_tiles((ks + 1) * 32);
    const int cur = ks & 1;
    const int ar = (rs * 16 + ln) * LROW;
    v16h a  = ld16(&la[cur][ar + 16 * grp], &la[cur][ar + 16 * grp + 8]);
    const int br0 = (cs * 16 + ln) * LROW;
    const int br1 = ((cs + 1) * 16 + ln) * LROW;
    v16h b0 = ld16(&lb[cur][br0 + 16 * grp], &lb[cur][br0 + 16 * grp + 8]);
    v16h b1 = ld16(&lb[cur][br1 + 16 * grp], &lb[cur][br1 + 16 * grp + 8]);
    c0 = WMMA_F16(a, b0, c0);
    c1 = WMMA_F16(a, b1, c1);
    if (ks + 1 < KSTEPS) store_tiles((ks + 1) & 1);
    __syncthreads();
  }

#pragma unroll
  for (int t = 0; t < 2; ++t) {
    v8f cc = t ? c1 : c0;
    int gc = n0 + (cs + t) * 16 + ln;
    int which = gc / DIM;           // 0=q 1=k 2=v
    int cin = gc % DIM;
    int h = cin / HD, d = cin % HD;
    float bias = (which == 0) ? qb[cin] : (which == 2 ? vb[cin] : 0.f);
#pragma unroll
    for (int r = 0; r < 8; ++r) {
      int gr = m0 + rs * 16 + r + 8 * grp;
      if (gr >= M_ROWS) continue;
      int b = gr / N_TOK, n = gr % N_TOK;
      float v = cc[r] + bias;
      if (which == 0) {
        v *= Q_SCALE;
        qws[(((size_t)(b * HEADS + h)) * NPAD + n) * HD + d] = (_Float16)v;
      } else if (which == 1) {
        kws[(((size_t)(b * HEADS + h)) * NPAD + n) * HD + d] = (_Float16)v;
      } else {
        vtws[(((size_t)(b * HEADS + h)) * HD + d) * NPAD + n] = (_Float16)v;
      }
    }
  }
}

// ---------------------------------------------------------------------------
// Kernel 2: flash-style attention per (b,h); one wave per 16-row Q tile.
// ---------------------------------------------------------------------------
__global__ void __launch_bounds__(256) attn_kernel(
    const _Float16* __restrict__ qws, const _Float16* __restrict__ kws,
    const _Float16* __restrict__ vtws, const float* __restrict__ rpb,
    const int* __restrict__ ridx, _Float16* __restrict__ aout) {
  __shared__ _Float16 lp[8 * 16 * LROW];   // per-wave P staging
  const int bh = blockIdx.x;
  const int b = bh / HEADS, h = bh % HEADS;
  const int wave = threadIdx.x >> 5, lane = threadIdx.x & 31;
  const int grp = lane >> 4, ln = lane & 15;
  const int qt = blockIdx.y * 8 + wave;
  if (qt >= 65) return;                 // ceil(1025/16)

  const _Float16* qbase = qws + (size_t)bh * NPAD * HD;
  const _Float16* kbase = kws + (size_t)bh * NPAD * HD;
  const _Float16* vbase = vtws + (size_t)bh * HD * NPAD;
  _Float16* lpw = lp + wave * (16 * LROW);

  const _Float16* qp = qbase + (size_t)(qt * 16 + ln) * HD;
  v16h qa0 = ld16(qp + 16 * grp, qp + 16 * grp + 8);        // d 0..31
  v16h qa1 = ld16(qp + 32 + 16 * grp, qp + 32 + 16 * grp + 8);   // d 32..63

  // hoisted per-row rpb row pointers (constant over kv loop)
  const int* rowp[8];
#pragma unroll
  for (int r = 0; r < 8; ++r) {
    int qq = qt * 16 + r + 8 * grp;
    int qqc = qq < N_TOK ? qq : N_TOK - 1;
    rowp[r] = ridx + (size_t)qqc * N_TOK;
  }

  v8f oacc[4] = {{}, {}, {}, {}};
  float mrow[8], lrow[8];
  const float NINF = -__builtin_inff();
#pragma unroll
  for (int r = 0; r < 8; ++r) { mrow[r] = NINF; lrow[r] = 0.f; }

  for (int kt = 0; kt < 33; ++kt) {
    const int key0 = kt * 32;
    const _Float16* kp0 = kbase + (size_t)(key0 + ln) * HD;
    const _Float16* kp1 = kbase + (size_t)(key0 + 16 + ln) * HD;
    v16h kb00 = ld16(kp0 + 16 * grp, kp0 + 16 * grp + 8);
    v16h kb01 = ld16(kp0 + 32 + 16 * grp, kp0 + 32 + 16 * grp + 8);
    v16h kb10 = ld16(kp1 + 16 * grp, kp1 + 16 * grp + 8);
    v16h kb11 = ld16(kp1 + 32 + 16 * grp, kp1 + 32 + 16 * grp + 8);

    // issue V fragment loads early: latency hides under QK^T + softmax
    v16h vbf[4];
#pragma unroll
    for (int dt = 0; dt < 4; ++dt) {
      const _Float16* vp = vbase + (size_t)(dt * 16 + ln) * NPAD + key0 + 16 * grp;
      vbf[dt] = ld16(vp, vp + 8);
    }

    v8f s0 = {}, s1 = {};
    s0 = WMMA_F16(qa0, kb00, s0);
    s0 = WMMA_F16(qa1, kb01, s0);
    s1 = WMMA_F16(qa0, kb10, s1);
    s1 = WMMA_F16(qa1, kb11, s1);

    const int kc0 = key0 + ln, kc1 = key0 + 16 + ln;
    const int kc0c = kc0 < N_TOK ? kc0 : N_TOK - 1;
    const int kc1c = kc1 < N_TOK ? kc1 : N_TOK - 1;
#pragma unroll
    for (int r = 0; r < 8; ++r) {
      float b0 = rpb[rowp[r][kc0c] * HEADS + h];   // branchless gather
      float b1 = rpb[rowp[r][kc1c] * HEADS + h];
      float v0 = (kc0 < N_TOK) ? s0[r] + b0 : NINF;
      float v1 = (kc1 < N_TOK) ? s1[r] + b1 : NINF;

      float mx = rmax16(fmaxf(v0, v1));
      float mn = fmaxf(mrow[r], mx);
      float corr = (mrow[r] == NINF) ? 0.f : __expf(mrow[r] - mn);
      float p0 = (v0 == NINF) ? 0.f : __expf(v0 - mn);
      float p1 = (v1 == NINF) ? 0.f : __expf(v1 - mn);
      lrow[r] = lrow[r] * corr + rsum16(p0 + p1);
      mrow[r] = mn;
      oacc[0][r] *= corr; oacc[1][r] *= corr;
      oacc[2][r] *= corr; oacc[3][r] *= corr;
      lpw[(r + 8 * grp) * LROW + ln] = (_Float16)p0;
      lpw[(r + 8 * grp) * LROW + 16 + ln] = (_Float16)p1;
    }
    asm volatile("s_wait_dscnt 0" ::: "memory");   // same-wave LDS RAW fence
    v16h pa = ld16(&lpw[ln * LROW + 16 * grp], &lpw[ln * LROW + 16 * grp + 8]);

#pragma unroll
    for (int dt = 0; dt < 4; ++dt)
      oacc[dt] = WMMA_F16(pa, vbf[dt], oacc[dt]);
  }

#pragma unroll
  for (int r = 0; r < 8; ++r) {
    const int qq = qt * 16 + r + 8 * grp;
    if (qq >= N_TOK) continue;
    float inv = (lrow[r] > 0.f) ? 1.f / lrow[r] : 0.f;
    size_t ob = ((size_t)(b * N_TOK + qq)) * DIM + h * HD;
#pragma unroll
    for (int dt = 0; dt < 4; ++dt)
      aout[ob + dt * 16 + ln] = (_Float16)(oacc[dt][r] * inv);
  }
}

// ---------------------------------------------------------------------------
// Kernel 3: out = attn_out @ proj_w^T + proj_b  (fp32 output)
// ---------------------------------------------------------------------------
__global__ void __launch_bounds__(256) proj_kernel(
    const _Float16* __restrict__ a, const float* __restrict__ w,
    const float* __restrict__ pb, float* __restrict__ out) {
  __shared__ _Float16 la[2][64 * LROW];
  __shared__ _Float16 lb[2][64 * LROW];
  const int m0 = blockIdx.x * 64;
  const int n0 = blockIdx.y * 64;
  const int tid = threadIdx.x;
  const int wave = tid >> 5, lane = tid & 31, grp = lane >> 4, ln = lane & 15;
  const int rs = wave >> 1;
  const int cs = (wave & 1) * 2;

  // A: one h8 (16B) per thread; B: two float4 per thread
  const int arA = tid >> 2, acA = (tid & 3) * 8;
  const int grA = m0 + arA;
  const size_t rowAa = (size_t)(grA < M_ROWS ? grA : M_ROWS - 1) * DIM;
  const int i0 = tid, i1 = tid + 256;
  const int br0i = i0 >> 3, bc0 = (i0 & 7) * 4;
  const int br1i = i1 >> 3, bc1 = (i1 & 7) * 4;
  const size_t rowB0 = (size_t)(n0 + br0i) * DIM;
  const size_t rowB1 = (size_t)(n0 + br1i) * DIM;

  h8 va; float4 fb0, fb1;
  auto load_tiles = [&](int kk) {
    va = *(const h8*)(a + rowAa + kk + acA);
    fb0 = *(const float4*)(w + rowB0 + kk + bc0);
    fb1 = *(const float4*)(w + rowB1 + kk + bc1);
    if (grA >= M_ROWS) { h8 z = {}; va = z; }
  };
  auto store_tiles = [&](int buf) {
    *(h8*)&la[buf][arA * LROW + acA] = va;
    *(h4*)&lb[buf][br0i * LROW + bc0] = cvt4(fb0);
    *(h4*)&lb[buf][br1i * LROW + bc1] = cvt4(fb1);
  };

  v8f c0 = {}, c1 = {};
  load_tiles(0);
  store_tiles(0);
  __syncthreads();
  for (int ks = 0; ks < KSTEPS; ++ks) {
    if (ks + 1 < KSTEPS) load_tiles((ks + 1) * 32);
    const int cur = ks & 1;
    const int ar = (rs * 16 + ln) * LROW;
    v16h av = ld16(&la[cur][ar + 16 * grp], &la[cur][ar + 16 * grp + 8]);
    const int br0 = (cs * 16 + ln) * LROW;
    const int br1 = ((cs + 1) * 16 + ln) * LROW;
    v16h b0 = ld16(&lb[cur][br0 + 16 * grp], &lb[cur][br0 + 16 * grp + 8]);
    v16h b1 = ld16(&lb[cur][br1 + 16 * grp], &lb[cur][br1 + 16 * grp + 8]);
    c0 = WMMA_F16(av, b0, c0);
    c1 = WMMA_F16(av, b1, c1);
    if (ks + 1 < KSTEPS) store_tiles((ks + 1) & 1);
    __syncthreads();
  }

#pragma unroll
  for (int t = 0; t < 2; ++t) {
    v8f cc = t ? c1 : c0;
    int gc = n0 + (cs + t) * 16 + ln;
    float bias = pb[gc];
#pragma unroll
    for (int r = 0; r < 8; ++r) {
      int gr = m0 + rs * 16 + r + 8 * grp;
      if (gr >= M_ROWS) continue;
      out[(size_t)gr * DIM + gc] = cc[r] + bias;
    }
  }
}

// ---------------------------------------------------------------------------
extern "C" void kernel_launch(void* const* d_in, const int* in_sizes, int n_in,
                              void* d_out, int out_size, void* d_ws, size_t ws_size,
                              hipStream_t stream) {
  (void)in_sizes; (void)n_in; (void)out_size; (void)ws_size;
  const float* x      = (const float*)d_in[0];
  const float* qkv_w  = (const float*)d_in[1];
  const float* q_bias = (const float*)d_in[2];
  const float* v_bias = (const float*)d_in[3];
  const float* rpb    = (const float*)d_in[4];
  const float* proj_w = (const float*)d_in[5];
  const float* proj_b = (const float*)d_in[6];
  const int*   ridx   = (const int*)d_in[7];

  char* ws = (char*)d_ws;
  const size_t ONE = (size_t)BATCH * HEADS * NPAD * HD * sizeof(unsigned short);
  _Float16* qws  = (_Float16*)(ws);
  _Float16* kws  = (_Float16*)(ws + ONE);
  _Float16* vtws = (_Float16*)(ws + 2 * ONE);
  _Float16* aout = (_Float16*)(ws + 3 * ONE);

  dim3 g1(129, 36);  // ceil(8200/64) x (2304/64)
  qkv_kernel<<<g1, 256, 0, stream>>>(x, qkv_w, q_bias, v_bias, qws, kws, vtws);

  dim3 g2(BATCH * HEADS, 9);  // 96 (b,h) x 9 chunks of 8 q-tiles
  attn_kernel<<<g2, 256, 0, stream>>>(qws, kws, vtws, rpb, ridx, aout);

  dim3 g3(129, 12);  // ceil(8200/64) x (768/64)
  proj_kernel<<<g3, 256, 0, stream>>>(aout, proj_w, proj_b, (float*)d_out);
}
